// Attention_27212912787927
// MI455X (gfx1250) — matrix operand order repaired and checked by reading, not compiled, not run on hardware
//
#include <hip/hip_runtime.h>

// ---------------------------------------------------------------------------
// CDNA5 (gfx1250) windowed attention with AdaLayerNorm modulation.
// bf16 WMMA (v_wmma_f32_16x16x32_bf16) with f32 accumulation throughout.
// ---------------------------------------------------------------------------

typedef __attribute__((ext_vector_type(16))) __bf16 v16bf;
typedef __attribute__((ext_vector_type(8)))  float  v8f;

#define NTOK   49
#define NPAD   64
#define DIMC   256
#define XS     264   // row stride (halves) for xm / q / k buffers (16B aligned)
#define VS     72    // row stride (halves) for V-transposed buffer
#define AS     72    // row stride (halves) for per-head attn buffer

union ABPack { uint4 q[2]; v16bf v; };

static __device__ __forceinline__ unsigned short f2bf(float f) {
  unsigned int u = __float_as_uint(f);
  u += 0x7fffu + ((u >> 16) & 1u);          // round-to-nearest-even
  return (unsigned short)(u >> 16);
}

static __device__ __forceinline__ v8f v8f_zero() {
  v8f z;
  #pragma unroll
  for (int i = 0; i < 8; ++i) z[i] = 0.0f;
  return z;
}

// A-operand (16x32 bf16): lane holds row m0+lsub; K pattern:
// lanes 0-15 -> K k0+0..15; lanes 16-31 -> K k0+16..31
static __device__ __forceinline__ v16bf load_a(const unsigned short* base, int m0,
                                               int k0, int stride, int lsub, int h) {
  const unsigned short* p = base + (m0 + lsub) * stride + k0 + 16 * h;
  ABPack t;
  t.q[0] = *(const uint4*)(p);
  t.q[1] = *(const uint4*)(p + 8);
  return t.v;
}

// B-operand (32x16 bf16) from a row-major [N][K] source (i.e. B = srcT):
// lane holds column n0+lsub of B; lanes 0-15 -> K k0+0..15, lanes 16-31 -> K k0+16..31
static __device__ __forceinline__ v16bf load_b(const unsigned short* base, int n0,
                                               int k0, int stride, int lsub, int h) {
  const unsigned short* p = base + (n0 + lsub) * stride + k0 + 16 * h;
  ABPack t;
  t.q[0] = *(const uint4*)(p);
  t.q[1] = *(const uint4*)(p + 8);
  return t.v;
}

// Pre-swizzled weight B-operand: [ntile][kc][lane][16] halves, fully contiguous.
static __device__ __forceinline__ v16bf load_bswz(const unsigned short* wbase,
                                                  int ntile, int kc, int lane) {
  const unsigned short* p = wbase + (((ntile * 8 + kc) * 32 + lane) << 4);
  ABPack t;
  t.q[0] = *(const uint4*)(p);
  t.q[1] = *(const uint4*)(p + 8);
  return t.v;
}

static __device__ __forceinline__ v8f wmma_bf16(v16bf a, v16bf b, v8f c) {
  return __builtin_amdgcn_wmma_f32_16x16x32_bf16(false, a, false, b,
                                                 (short)0, c, false, false);
}

// ---------------------------------------------------------------------------
// Prologue 1: gbs[b][o] = emb[b] . proj_w[o]   ([8,768] <- [8,512]x[768,512])
// ---------------------------------------------------------------------------
__global__ void gbs_kernel(const float* __restrict__ emb,
                           const float* __restrict__ proj_w,
                           float* __restrict__ gbs) {
  int idx = blockIdx.x * blockDim.x + threadIdx.x;
  if (idx >= 8 * 768) return;
  int b = idx / 768, o = idx % 768;
  const float* e = emb + b * 512;
  const float* w = proj_w + o * 512;
  float s = 0.0f;
  #pragma unroll 4
  for (int i = 0; i < 512; ++i) s = fmaf(e[i], w[i], s);
  gbs[idx] = s;
}

// ---------------------------------------------------------------------------
// Prologue 2: convert wq/wk/wv/wo to bf16, pre-swizzled into WMMA B layout.
// Softmax scale 1/sqrt(32) folded into wq.
// ---------------------------------------------------------------------------
__global__ void wswz_kernel(const float* __restrict__ wq, const float* __restrict__ wk,
                            const float* __restrict__ wv, const float* __restrict__ wo,
                            unsigned short* __restrict__ wswz) {
  int idx = blockIdx.x * blockDim.x + threadIdx.x;   // 4 * 65536
  if (idx >= 4 * 65536) return;
  int mat   = idx >> 16;
  int r     = idx & 65535;
  int e     = r & 15;
  int lane  = (r >> 4) & 31;
  int kc    = (r >> 9) & 7;
  int ntile = (r >> 12) & 15;
  int lsub = lane & 15, hh = lane >> 4;
  int n = ntile * 16 + lsub;
  int k = kc * 32 + 16 * hh + e;
  const float* W = (mat == 0) ? wq : (mat == 1) ? wk : (mat == 2) ? wv : wo;
  float v = W[n * 256 + k];
  if (mat == 0) v *= 0.17677669529663688f;   // DH^-0.5
  wswz[idx] = f2bf(v);
}

// ---------------------------------------------------------------------------
// Main kernel: one workgroup (8 wave32) per 7x7 window.
// ---------------------------------------------------------------------------
__global__ __launch_bounds__(256) void attn_win_kernel(
    const float* __restrict__ x, const float* __restrict__ ln_w,
    const float* __restrict__ ln_b, const float* __restrict__ bias_table,
    const float* __restrict__ gbs, const unsigned short* __restrict__ wswz,
    float* __restrict__ out) {

  __shared__ unsigned short s_xm[NPAD * XS];           // xm, later attn-out (bf16)
  __shared__ unsigned short s_qka[8 * NPAD * AS];      // union: {Q,K} then per-head attn
  __shared__ unsigned short s_vT[DIMC * VS];           // V transposed [channel][token]
  __shared__ float          s_mod[3 * DIMC];           // gamma | beta | sigma
  __shared__ float          s_bias[169 * 8];           // relative position bias table

  const int win  = blockIdx.x;          // 0..2047
  const int bat  = win >> 8;            // batch index (win / (16*16))
  const int tid  = threadIdx.x;
  const int wave = tid >> 5;
  const int lane = tid & 31;
  const int lsub = lane & 15;
  const int h    = lane >> 4;

  // ---- load modulation vectors + bias table --------------------------------
  for (int i = tid; i < 3 * DIMC; i += 256) s_mod[i] = gbs[bat * 768 + i];
  for (int i = tid; i < 169 * 8;  i += 256) s_bias[i] = bias_table[i];
  __syncthreads();

  // ---- stage 1: LayerNorm + AdaLN modulation -> bf16 xm in LDS -------------
  const float* xwin = x + (size_t)win * (NTOK * DIMC);
  for (int row = wave; row < NPAD; row += 8) {
    if (row < NTOK) {
      const float* xr = xwin + row * DIMC;
      float v[8], s = 0.0f, ss = 0.0f;
      #pragma unroll
      for (int j = 0; j < 8; ++j) {
        v[j] = xr[lane + 32 * j];
        s += v[j]; ss += v[j] * v[j];
      }
      #pragma unroll
      for (int off = 16; off >= 1; off >>= 1) {
        s  += __shfl_xor(s, off, 32);
        ss += __shfl_xor(ss, off, 32);
      }
      float mean = s * (1.0f / 256.0f);
      float var  = ss * (1.0f / 256.0f) - mean * mean;
      float rstd = rsqrtf(var + 1e-5f);
      #pragma unroll
      for (int j = 0; j < 8; ++j) {
        int c = lane + 32 * j;
        float xn = (v[j] - mean) * rstd * ln_w[c] + ln_b[c];
        float xm = xn * (1.0f + s_mod[c]) + s_mod[DIMC + c];
        s_xm[row * XS + c] = f2bf(xm);
      }
    } else {
      #pragma unroll
      for (int j = 0; j < 8; ++j) s_xm[row * XS + lane + 32 * j] = 0;
    }
  }
  __syncthreads();

  // ---- stage 2: Q/K/V projections (192 16x16 tiles over 8 waves) -----------
  unsigned short* sq = s_qka;
  unsigned short* sk = s_qka + NPAD * XS;
  for (int t = wave; t < 192; t += 8) {
    int mat = t >> 6;                  // 0=Q 1=K 2=V
    int rt  = t & 63;
    int tm  = rt & 3, tn = rt >> 2;    // m-tile 0..3, n-tile 0..15
    const unsigned short* wb = wswz + mat * 65536;
    v8f acc = v8f_zero();
    #pragma unroll
    for (int kc = 0; kc < 8; ++kc) {
      v16bf a = load_a(s_xm, tm * 16, kc * 32, XS, lsub, h);
      v16bf b = load_bswz(wb, tn, kc, lane);
      acc = wmma_bf16(a, b, acc);
    }
    if (mat == 2) {                    // V stored transposed [channel][token]
      #pragma unroll
      for (int r = 0; r < 8; ++r)
        s_vT[(tn * 16 + lsub) * VS + tm * 16 + r + 8 * h] = f2bf(acc[r]);
    } else {
      unsigned short* dst = (mat == 0) ? sq : sk;
      #pragma unroll
      for (int r = 0; r < 8; ++r)
        dst[(tm * 16 + r + 8 * h) * XS + tn * 16 + lsub] = f2bf(acc[r]);
    }
  }
  __syncthreads();

  // ---- stage 3: attention, one head per wave -------------------------------
  const int cbase = wave * 32;   // head channel offset
  v8f sim[4][4];
  #pragma unroll
  for (int tm = 0; tm < 4; ++tm) {
    v16bf a = load_a(sq, tm * 16, cbase, XS, lsub, h);
    #pragma unroll
    for (int tj = 0; tj < 4; ++tj) {
      v16bf b = load_b(sk, tj * 16, cbase, XS, lsub, h);   // B = K^T
      if (tm == 0) sim[tm][tj] = v8f_zero(); else sim[tm][tj] = v8f_zero();
      sim[tm][tj] = wmma_bf16(a, b, sim[tm][tj]);
    }
  }
  __syncthreads();   // all waves finished reading Q/K; region reused for attn

  unsigned short* sattn = s_qka + wave * NPAD * AS;
  #pragma unroll
  for (int tm = 0; tm < 4; ++tm) {
    #pragma unroll
    for (int r = 0; r < 8; ++r) {
      int i = tm * 16 + r + 8 * h;
      float rowv[4];
      float m = -3.0e38f;
      #pragma unroll
      for (int tj = 0; tj < 4; ++tj) {
        int j = tj * 16 + lsub;
        float sv = sim[tm][tj][r];
        if (j < NTOK) {
          if (i < NTOK) {
            int di = i / 7 - j / 7 + 6;
            int dj = i % 7 - j % 7 + 6;
            sv += s_bias[(di * 13 + dj) * 8 + wave];
          }
        } else {
          sv = -3.0e38f;               // mask padded keys
        }
        rowv[tj] = sv;
        m = fmaxf(m, sv);
      }
      #pragma unroll
      for (int off = 8; off >= 1; off >>= 1) m = fmaxf(m, __shfl_xor(m, off, 32));
      float ssum = 0.0f;
      #pragma unroll
      for (int tj = 0; tj < 4; ++tj) { rowv[tj] = __expf(rowv[tj] - m); ssum += rowv[tj]; }
      #pragma unroll
      for (int off = 8; off >= 1; off >>= 1) ssum += __shfl_xor(ssum, off, 32);
      float inv = 1.0f / ssum;
      #pragma unroll
      for (int tj = 0; tj < 4; ++tj)
        sattn[i * AS + tj * 16 + lsub] = f2bf(rowv[tj] * inv);
    }
  }

  // attn @ V  ->  write into s_xm (free since stage 2)
  #pragma unroll
  for (int tm = 0; tm < 4; ++tm) {
    #pragma unroll
    for (int tn = 0; tn < 2; ++tn) {
      v8f acc = v8f_zero();
      #pragma unroll
      for (int kc = 0; kc < 2; ++kc) {
        v16bf a = load_a(sattn, tm * 16, kc * 32, AS, lsub, h);
        v16bf b = load_b(s_vT, cbase + tn * 16, kc * 32, VS, lsub, h);
        acc = wmma_bf16(a, b, acc);
      }
      #pragma unroll
      for (int r = 0; r < 8; ++r)
        s_xm[(tm * 16 + r + 8 * h) * XS + cbase + tn * 16 + lsub] = f2bf(acc[r]);
    }
  }
  __syncthreads();

  // ---- stage 4: O projection + sigma gate + store --------------------------
  const unsigned short* wo_b = wswz + 3 * 65536;
  float* owin = out + (size_t)win * (NTOK * DIMC);
  for (int t = wave; t < 64; t += 8) {
    int tm = t & 3, tn = t >> 2;
    v8f acc = v8f_zero();
    #pragma unroll
    for (int kc = 0; kc < 8; ++kc) {
      v16bf a = load_a(s_xm, tm * 16, kc * 32, XS, lsub, h);
      v16bf b = load_bswz(wo_b, tn, kc, lane);
      acc = wmma_bf16(a, b, acc);
    }
    #pragma unroll
    for (int r = 0; r < 8; ++r) {
      int tok = tm * 16 + r + 8 * h;
      int c   = tn * 16 + lsub;
      if (tok < NTOK)
        owin[tok * DIMC + c] = acc[r] * s_mod[2 * DIMC + c];
    }
  }
}

// ---------------------------------------------------------------------------
extern "C" void kernel_launch(void* const* d_in, const int* in_sizes, int n_in,
                              void* d_out, int out_size, void* d_ws, size_t ws_size,
                              hipStream_t stream) {
  const float* x          = (const float*)d_in[0];
  const float* emb        = (const float*)d_in[1];
  const float* ln_w       = (const float*)d_in[2];
  const float* ln_b       = (const float*)d_in[3];
  const float* proj_w     = (const float*)d_in[4];
  const float* wq         = (const float*)d_in[5];
  const float* wk         = (const float*)d_in[6];
  const float* wv         = (const float*)d_in[7];
  const float* wo         = (const float*)d_in[8];
  const float* bias_table = (const float*)d_in[9];
  float* out = (float*)d_out;

  float* gbs = (float*)d_ws;                                    // 8*768 f32
  unsigned short* wswz = (unsigned short*)((char*)d_ws + 8 * 768 * sizeof(float));

  gbs_kernel<<<24, 256, 0, stream>>>(emb, proj_w, gbs);
  wswz_kernel<<<1024, 256, 0, stream>>>(wq, wk, wv, wo, wswz);
  attn_win_kernel<<<2048, 256, 0, stream>>>(x, ln_w, ln_b, bias_table, gbs, wswz, out);
}
